// TorchModel_53257594470869
// MI455X (gfx1250) — matrix operand order repaired and checked by reading, not compiled, not run on hardware
//
#include <hip/hip_runtime.h>
#include <hip/hip_bf16.h>
#include <math.h>

typedef __bf16 bf16_t;
typedef __attribute__((ext_vector_type(16))) __bf16 v16bf;
typedef __attribute__((ext_vector_type(8)))  __bf16 v8bf;
typedef __attribute__((ext_vector_type(8)))  float  v8f;

#define NN 50000      // nodes
#define NE 500000     // edges
#define NF 128        // node features
#define EF 16         // edge features

__device__ __forceinline__ float gelu_exact(float x) {
    return 0.5f * x * (1.0f + erff(x * 0.70710678118654752f));
}

// ---------------------------------------------------------------------------
// Generic WMMA GEMM:  out = act( A[M,K](bf16) @ W[K,N](bf16) + Cin + bias )
// W is stored TRANSPOSED (WT[Np, Kp], row-major, ldw = Kp) so each lane's
// B fragment (two runs of 8 consecutive K for one column) is two contiguous 16B runs.
// One 16x16 output tile per wave32; 4 waves / 128-thread block.
// Fragment layouts follow CDNA5 ISA 7.12.2 (16-bit A 16x32, B 32x16, f32 C/D).
// All fragment chunks are 16B-aligned -> global_load_b128.
// ---------------------------------------------------------------------------
__global__ __launch_bounds__(128)
void wmma_gemm(const bf16_t* __restrict__ A, int lda,
               const bf16_t* __restrict__ WT, int ldw,
               const float* __restrict__ bias,
               const float* __restrict__ Cin, int ldc,
               float* __restrict__ outF, int ldof,
               bf16_t* __restrict__ outH, int ldoh,
               int M, int N, int K, int Nreal, int act)
{
    const int wave = threadIdx.x >> 5;
    const int lane = threadIdx.x & 31;
    const int tilesN = N >> 4;
    const long total = (long)(M >> 4) * tilesN;
    const long tile  = (long)blockIdx.x * 4 + wave;
    if (tile >= total) return;                 // wave-uniform: EXEC stays all-ones
    const int tm = (int)(tile / tilesN);
    const int tn = (int)(tile % tilesN);

    const int l16  = lane & 15;
    const int hi   = lane >> 4;                // 0 for lanes 0-15, 1 for 16-31
    const int row  = (tm << 4) + l16;          // A row for this lane
    const int colg = (tn << 4) + l16;          // B/C column for this lane
    const int kbA  = hi ? 8  : 0;              // A fragment K-base (ISA table)
    const int kbB  = hi ? 8  : 0;              // B fragment K-base (ISA table)
    const int rbase = (tm << 4) + (hi << 3);   // C/D rows: VGPR r -> rbase + r

    const bf16_t* Abase = A  + (size_t)row  * lda + kbA;
    const bf16_t* Bbase = WT + (size_t)colg * ldw + kbB;

    v8f c = {0.f,0.f,0.f,0.f,0.f,0.f,0.f,0.f};
    if (Cin) {
#pragma unroll
        for (int r = 0; r < 8; ++r)
            c[r] = Cin[(size_t)(rbase + r) * ldc + colg];
    }

    for (int k0 = 0; k0 < K; k0 += 32) {
        v8bf alo = *(const v8bf*)(Abase + k0);        // K = kbA + 0..7
        v8bf ahi = *(const v8bf*)(Abase + k0 + 16);   // K = kbA + 16..23
        v8bf blo = *(const v8bf*)(Bbase + k0);        // K = kbB + 0..7
        v8bf bhi = *(const v8bf*)(Bbase + k0 + 16);   // K = kbB + 16..23
        v16bf a = __builtin_shufflevector(alo, ahi,
                      0,1,2,3,4,5,6,7,8,9,10,11,12,13,14,15);
        v16bf b = __builtin_shufflevector(blo, bhi,
                      0,1,2,3,4,5,6,7,8,9,10,11,12,13,14,15);
        c = __builtin_amdgcn_wmma_f32_16x16x32_bf16(
                false, a, false, b, (short)0, c, false, false);
    }

    const bool colok = colg < Nreal;
    const float bv = bias ? bias[colg] : 0.f;
#pragma unroll
    for (int r = 0; r < 8; ++r) {
        float v = c[r] + bv;
        if (act == 1) v = gelu_exact(v);
        else if (act == 2) v = 1.f / (1.f + expf(-v));
        if (colok) {
            if (outF) outF[(size_t)(rbase + r) * ldof + colg] = v;
            if (outH) outH[(size_t)(rbase + r) * ldoh + colg] = (bf16_t)v;
        }
    }
}

// ---------------------------------------------------------------------------
// Support kernels
// ---------------------------------------------------------------------------
__global__ void fill0(float* p, size_t n) {
    size_t i = (size_t)blockIdx.x * blockDim.x + threadIdx.x;
    for (; i < n; i += (size_t)gridDim.x * blockDim.x) p[i] = 0.f;
}

__global__ void degree_k(const int* __restrict__ dst, float* deg) {
    int e = blockIdx.x * blockDim.x + threadIdx.x;
    if (e < NE) atomicAdd(&deg[dst[e]], 1.f);
}
__global__ void dis_k(float* deg) {
    int i = blockIdx.x * blockDim.x + threadIdx.x;
    if (i < NN) { float d = deg[i]; deg[i] = d > 0.f ? rsqrtf(fmaxf(d, 1.f)) : 0.f; }
}
__global__ void norm_k(const int* __restrict__ src, const int* __restrict__ dst,
                       const float* __restrict__ dis, float* nrm) {
    int e = blockIdx.x * blockDim.x + threadIdx.x;
    if (e < NE) nrm[e] = dis[src[e]] * dis[dst[e]];
}

// column mean/var (biased), one block per column
__global__ void col_stats(const float* __restrict__ X, int rows, int cols,
                          float* mean, float* var) {
    __shared__ float s1[256], s2[256];
    int cidx = blockIdx.x;
    float s = 0.f, sq = 0.f;
    for (int r = threadIdx.x; r < rows; r += 256) {
        float v = X[(size_t)r * cols + cidx];
        s += v; sq += v * v;
    }
    s1[threadIdx.x] = s; s2[threadIdx.x] = sq; __syncthreads();
    for (int o = 128; o > 0; o >>= 1) {
        if ((int)threadIdx.x < o) { s1[threadIdx.x] += s1[threadIdx.x + o];
                                    s2[threadIdx.x] += s2[threadIdx.x + o]; }
        __syncthreads();
    }
    if (threadIdx.x == 0) {
        float mu = s1[0] / rows;
        mean[cidx] = mu;
        var[cidx]  = s2[0] / rows - mu * mu;
    }
}

__global__ void bn_cast_node(const float* __restrict__ X, const float* mean,
                             const float* var, const float* g, const float* b,
                             bf16_t* xn, bf16_t* xraw) {
    size_t idx = (size_t)blockIdx.x * blockDim.x + threadIdx.x;
    if (idx >= (size_t)NN * NF) return;
    int cc = (int)(idx & (NF - 1));
    float x = X[idx];
    xraw[idx] = (bf16_t)x;
    float v = (x - mean[cc]) * rsqrtf(var[cc] + 1e-5f) * g[cc] + b[cc];
    xn[idx] = (bf16_t)v;
}

// TAGConv hop: acc[dst] += h[src] * norm  (256 features)
// One thread per (edge, 8-feature chunk): single b128 bf16 gather + 8 atomics.
__global__ void propagate256(const bf16_t* __restrict__ h, const int* __restrict__ src,
                             const int* __restrict__ dst, const float* __restrict__ nrm,
                             float* __restrict__ acc) {
    size_t idx = (size_t)blockIdx.x * blockDim.x + threadIdx.x;
    size_t e = idx >> 5; int c8 = (int)(idx & 31) << 3;
    if (e >= NE) return;
    int s = src[e], d = dst[e];
    float w = nrm[e];
    v8bf hv = *(const v8bf*)(h + (size_t)s * 256 + c8);
    float* ap = acc + (size_t)d * 256 + c8;
#pragma unroll
    for (int i = 0; i < 8; ++i) atomicAdd(ap + i, (float)hv[i] * w);
}

__global__ void cast_f2b(const float* __restrict__ in, bf16_t* __restrict__ out, size_t n) {
    size_t i = ((size_t)blockIdx.x * blockDim.x + threadIdx.x) * 4;
    if (i >= n) return;
    float4 v = *(const float4*)(in + i);
    bf16_t o[4] = {(bf16_t)v.x, (bf16_t)v.y, (bf16_t)v.z, (bf16_t)v.w};
    *(ulong1*)(out + i) = *(ulong1*)o;
}

__global__ void concat2(const bf16_t* __restrict__ a, const bf16_t* __restrict__ b,
                        bf16_t* __restrict__ out, size_t rows, int c1, int c2) {
    int ct = c1 + c2;
    size_t idx = (size_t)blockIdx.x * blockDim.x + threadIdx.x;
    if (idx >= rows * ct) return;
    size_t r = idx / ct; int cc = (int)(idx % ct);
    out[idx] = (cc < c1) ? a[r * c1 + cc] : b[r * c2 + (cc - c1)];
}

// column stats of eg_in = [ns | nd | edge_raw] without materializing it
__global__ void edge_bn_stats(const float* __restrict__ nemb, const float* __restrict__ eraw,
                              const int* __restrict__ src, const int* __restrict__ dst,
                              float* mean, float* var) {
    __shared__ float s1[256], s2[256];
    int cidx = blockIdx.x;   // 0..271
    float s = 0.f, sq = 0.f;
    for (int e = threadIdx.x; e < NE; e += 256) {
        float v;
        if (cidx < 128)      v = nemb[(size_t)src[e] * NF + cidx];
        else if (cidx < 256) v = nemb[(size_t)dst[e] * NF + (cidx - 128)];
        else                 v = eraw[(size_t)e * EF + (cidx - 256)];
        s += v; sq += v * v;
    }
    s1[threadIdx.x] = s; s2[threadIdx.x] = sq; __syncthreads();
    for (int o = 128; o > 0; o >>= 1) {
        if ((int)threadIdx.x < o) { s1[threadIdx.x] += s1[threadIdx.x + o];
                                    s2[threadIdx.x] += s2[threadIdx.x + o]; }
        __syncthreads();
    }
    if (threadIdx.x == 0) {
        float mu = s1[0] / NE;
        mean[cidx] = mu;
        var[cidx]  = s2[0] / NE - mu * mu;
    }
}

// build BN(concat[ns,nd,eraw]) -> bf16, K padded 272->288
__global__ void build_egin(const float* __restrict__ nemb, const float* __restrict__ eraw,
                           const int* __restrict__ src, const int* __restrict__ dst,
                           const float* mean, const float* var, const float* g,
                           const float* b, bf16_t* __restrict__ out) {
    size_t idx = (size_t)blockIdx.x * blockDim.x + threadIdx.x;
    if (idx >= (size_t)NE * 288) return;
    size_t e = idx / 288; int cc = (int)(idx % 288);
    float v = 0.f;
    if (cc < 272) {
        float x;
        if (cc < 128)      x = nemb[(size_t)src[e] * NF + cc];
        else if (cc < 256) x = nemb[(size_t)dst[e] * NF + (cc - 128)];
        else               x = eraw[(size_t)e * EF + (cc - 256)];
        v = (x - mean[cc]) * rsqrtf(var[cc] + 1e-5f) * g[cc] + b[cc];
    }
    out[idx] = (bf16_t)v;
}

// build concat[ns,nd,edge_emb,eraw] -> bf16, K padded 336->352
__global__ void build_mc(const bf16_t* __restrict__ nemb, const bf16_t* __restrict__ eemb,
                         const float* __restrict__ eraw, const int* __restrict__ src,
                         const int* __restrict__ dst, bf16_t* __restrict__ out) {
    size_t idx = (size_t)blockIdx.x * blockDim.x + threadIdx.x;
    if (idx >= (size_t)NE * 352) return;
    size_t e = idx / 352; int cc = (int)(idx % 352);
    bf16_t v = (bf16_t)0.f;
    if (cc < 128)      v = nemb[(size_t)src[e] * NF + cc];
    else if (cc < 256) v = nemb[(size_t)dst[e] * NF + (cc - 128)];
    else if (cc < 320) v = eemb[(size_t)e * 64 + (cc - 256)];
    else if (cc < 336) v = (bf16_t)eraw[(size_t)e * EF + (cc - 320)];
    out[idx] = v;
}

__global__ void softmax8(const float* __restrict__ lg, float* __restrict__ out) {
    int e = blockIdx.x * blockDim.x + threadIdx.x;
    if (e >= NE) return;
    float v[8], mx = -1e30f;
#pragma unroll
    for (int i = 0; i < 8; ++i) { v[i] = lg[(size_t)e * 8 + i]; mx = fmaxf(mx, v[i]); }
    float s = 0.f;
#pragma unroll
    for (int i = 0; i < 8; ++i) { v[i] = expf(v[i] - mx); s += v[i]; }
    float inv = 1.f / s;
#pragma unroll
    for (int i = 0; i < 8; ++i) out[(size_t)e * 8 + i] = v[i] * inv;
}

// fp32 W[K,N] -> bf16 transposed WT[Np, Kp] (zero-padded)
__global__ void pack_weight_t(const float* __restrict__ W, bf16_t* __restrict__ out,
                              int K, int N, int Kp, int Np) {
    int idx = blockIdx.x * blockDim.x + threadIdx.x;
    if (idx >= Kp * Np) return;
    int np = idx / Kp, kp = idx % Kp;
    out[idx] = (bf16_t)((kp < K && np < N) ? W[(size_t)kp * N + np] : 0.f);
}

__global__ void pad_bias(const float* __restrict__ b, float* __restrict__ out, int N, int Np) {
    int i = blockIdx.x * blockDim.x + threadIdx.x;
    if (i < Np) out[i] = (i < N) ? b[i] : 0.f;
}

// ---------------------------------------------------------------------------
extern "C" void kernel_launch(void* const* d_in, const int* in_sizes, int n_in,
                              void* d_out, int out_size, void* d_ws, size_t ws_size,
                              hipStream_t stream) {
    const float* node_x   = (const float*)d_in[0];
    const float* edge_raw = (const float*)d_in[1];
    const int*   src      = (const int*)d_in[2];
    const int*   dst      = src + NE;
    const float *n_bn_g = (const float*)d_in[3],  *n_bn_b = (const float*)d_in[4];
    const float *W1 = (const float*)d_in[5],  *b1 = (const float*)d_in[6];
    const float *T1 = (const float*)d_in[7],  *bT1 = (const float*)d_in[8];
    const float *W2 = (const float*)d_in[9],  *b2 = (const float*)d_in[10];
    const float *T2 = (const float*)d_in[11], *bT2 = (const float*)d_in[12];
    const float *W3 = (const float*)d_in[13], *b3 = (const float*)d_in[14];
    const float *W4 = (const float*)d_in[15], *b4 = (const float*)d_in[16];
    const float *e_bn_g = (const float*)d_in[17], *e_bn_b = (const float*)d_in[18];
    const float *We1 = (const float*)d_in[19], *be1 = (const float*)d_in[20];
    const float *We2 = (const float*)d_in[21], *be2 = (const float*)d_in[22];
    const float *Wc1 = (const float*)d_in[23], *bc1 = (const float*)d_in[24];
    const float *Wc2 = (const float*)d_in[25], *bc2 = (const float*)d_in[26];
    const float *Wcl = (const float*)d_in[27], *bcl = (const float*)d_in[28];
    const float *Wb1 = (const float*)d_in[29], *bb1 = (const float*)d_in[30];
    const float *Wb2 = (n_in > 31) ? (const float*)d_in[31] : nullptr;
    const float *bb2 = (n_in > 32) ? (const float*)d_in[32] : nullptr;
    if (!Wb2) { Wb2 = Wb1; bb2 = bb1; }

    float* outNemb = (float*)d_out;                       // [50000,128]
    float* outEemb = outNemb + (size_t)NN * NF;           // [500000,64]
    float* outMC   = outEemb + (size_t)NE * 64;           // [500000,8]
    float* outBin  = outMC   + (size_t)NE * 8;            // [500000]

    // ---- workspace arena ----
    char* ws = (char*)d_ws; size_t off = 0;
    auto alloc = [&](size_t bytes) -> void* {
        void* p = ws + off; off = (off + bytes + 255) & ~(size_t)255; return p;
    };
    float* dis   = (float*)alloc(NN * 4);
    float* nrm   = (float*)alloc((size_t)NE * 4);
    float* meanN = (float*)alloc(128 * 4); float* varN = (float*)alloc(128 * 4);
    float* meanE = (float*)alloc(272 * 4); float* varE = (float*)alloc(272 * 4);
    bf16_t* pW1  = (bf16_t*)alloc(128 * 256 * 2);
    bf16_t* pT1  = (bf16_t*)alloc(4 * 256 * 128 * 2);
    bf16_t* pW2  = (bf16_t*)alloc(128 * 256 * 2);
    bf16_t* pT2  = (bf16_t*)alloc(4 * 256 * 128 * 2);
    bf16_t* pW3  = (bf16_t*)alloc(256 * 128 * 2);
    bf16_t* pW4  = (bf16_t*)alloc(128 * 128 * 2);
    bf16_t* pWe1 = (bf16_t*)alloc(288 * 128 * 2);
    bf16_t* pWe2 = (bf16_t*)alloc(128 * 64 * 2);
    bf16_t* pWc1 = (bf16_t*)alloc(352 * 64 * 2);
    bf16_t* pWc2 = (bf16_t*)alloc(64 * 32 * 2);
    bf16_t* pWcl = (bf16_t*)alloc(32 * 16 * 2);
    bf16_t* pWb1 = (bf16_t*)alloc(64 * 64 * 2);
    bf16_t* pWb2 = (bf16_t*)alloc(64 * 16 * 2);
    float* pbcl  = (float*)alloc(16 * 4);
    float* pbb2  = (float*)alloc(16 * 4);
    bf16_t* xn   = (bf16_t*)alloc((size_t)NN * 128 * 2);
    bf16_t* xraw = (bf16_t*)alloc((size_t)NN * 128 * 2);
    bf16_t* hA   = (bf16_t*)alloc((size_t)NN * 256 * 2);
    bf16_t* hB   = (bf16_t*)alloc((size_t)NN * 256 * 2);
    bf16_t* h128 = (bf16_t*)alloc((size_t)NN * 128 * 2);
    bf16_t* catb = (bf16_t*)alloc((size_t)NN * 256 * 2);
    bf16_t* a_bf = (bf16_t*)alloc((size_t)NN * 128 * 2);
    bf16_t* nembB= (bf16_t*)alloc((size_t)NN * 128 * 2);
    float* accP  = (float*)alloc((size_t)NN * 256 * 4);
    float* accT  = (float*)alloc((size_t)NN * 128 * 4);
    bf16_t* egmc = (bf16_t*)alloc((size_t)NE * 352 * 2);   // eg_in(288) then mc(352)
    bf16_t* eh   = (bf16_t*)alloc((size_t)NE * 128 * 2);   // eh, then ch1 | bh
    bf16_t* eembB= (bf16_t*)alloc((size_t)NE * 64 * 2);
    bf16_t* ch2  = (bf16_t*)alloc((size_t)NE * 32 * 2);
    float* logit = (float*)alloc((size_t)NE * 8 * 4);
    bf16_t* ch1 = eh;
    bf16_t* bh  = eh + (size_t)NE * 64;

    // ldw passed to wmma_gemm is the TRANSPOSED leading dim = padded K
    auto gemm = [&](const bf16_t* A, int lda, const bf16_t* Wb, int ldw,
                    const float* bias, const float* Cin,
                    float* oF, int ldof, bf16_t* oH, int ldoh,
                    int M, int N, int K, int Nreal, int act) {
        long tiles = (long)(M / 16) * (N / 16);
        int grid = (int)((tiles + 3) / 4);
        wmma_gemm<<<grid, 128, 0, stream>>>(A, lda, Wb, ldw, bias, Cin, N,
                                            oF, ldof, oH, ldoh, M, N, K, Nreal, act);
    };
    auto pack = [&](const float* Wf, bf16_t* dp, int K, int N, int Kp, int Np) {
        int n = Kp * Np;
        pack_weight_t<<<(n + 255) / 256, 256, 0, stream>>>(Wf, dp, K, N, Kp, Np);
    };

    // ---- stage 0: weight packing (fp32 -> bf16 transposed, zero-padded) ----
    pack(W1, pW1, 128, 256, 128, 256);
    for (int k = 0; k < 4; ++k) pack(T1 + k * 256 * 128, pT1 + k * 256 * 128, 256, 128, 256, 128);
    pack(W2, pW2, 128, 256, 128, 256);
    for (int k = 0; k < 4; ++k) pack(T2 + k * 256 * 128, pT2 + k * 256 * 128, 256, 128, 256, 128);
    pack(W3, pW3, 256, 128, 256, 128);
    pack(W4, pW4, 128, 128, 128, 128);
    pack(We1, pWe1, 272, 128, 288, 128);
    pack(We2, pWe2, 128, 64, 128, 64);
    pack(Wc1, pWc1, 336, 64, 352, 64);
    pack(Wc2, pWc2, 64, 32, 64, 32);
    pack(Wcl, pWcl, 32, 8, 32, 16);
    pack(Wb1, pWb1, 64, 64, 64, 64);
    pack(Wb2, pWb2, 64, 1, 64, 16);
    pad_bias<<<1, 16, 0, stream>>>(bcl, pbcl, 8, 16);
    pad_bias<<<1, 16, 0, stream>>>(bb2, pbb2, 1, 16);

    // ---- stage 1: GCN normalization ----
    fill0<<<256, 256, 0, stream>>>(dis, NN);
    degree_k<<<(NE + 255) / 256, 256, 0, stream>>>(dst, dis);
    dis_k<<<(NN + 255) / 256, 256, 0, stream>>>(dis);
    norm_k<<<(NE + 255) / 256, 256, 0, stream>>>(src, dst, dis, nrm);

    // ---- stage 2: node BatchNorm + bf16 casts ----
    col_stats<<<128, 256, 0, stream>>>(node_x, NN, NF, meanN, varN);
    {
        size_t n = (size_t)NN * NF;
        bn_cast_node<<<(unsigned)((n + 255) / 256), 256, 0, stream>>>(
            node_x, meanN, varN, n_bn_g, n_bn_b, xn, xraw);
    }

    // ---- stage 3: NodeGLAM ----
    gemm(xn, 128, pW1, 128, b1, nullptr, nullptr, 0, hA, 256, NN, 256, 128, 256, 1);
    // TAGConv 1
    gemm(hA, 256, pT1, 256, nullptr, nullptr, accT, 128, nullptr, 0, NN, 128, 256, 128, 0);
    {
        bf16_t* cur = hA;
        for (int k = 1; k <= 3; ++k) {
            bf16_t* nxt = (cur == hA) ? hB : hA;
            fill0<<<2048, 256, 0, stream>>>(accP, (size_t)NN * 256);
            propagate256<<<(unsigned)(((size_t)NE * 32) / 256), 256, 0, stream>>>(
                cur, src, dst, nrm, accP);
            cast_f2b<<<(unsigned)(((size_t)NN * 256 / 4 + 255) / 256), 256, 0, stream>>>(
                accP, nxt, (size_t)NN * 256);
            bool last = (k == 3);
            gemm(nxt, 256, pT1 + k * 256 * 128, 256, last ? bT1 : nullptr, accT,
                 last ? nullptr : accT, 128, last ? h128 : nullptr, 128,
                 NN, 128, 256, 128, last ? 1 : 0);
            cur = nxt;
        }
    }
    gemm(h128, 128, pW2, 128, b2, nullptr, nullptr, 0, hA, 256, NN, 256, 128, 256, 1);
    // TAGConv 2
    gemm(hA, 256, pT2, 256, nullptr, nullptr, accT, 128, nullptr, 0, NN, 128, 256, 128, 0);
    {
        bf16_t* cur = hA;
        for (int k = 1; k <= 3; ++k) {
            bf16_t* nxt = (cur == hA) ? hB : hA;
            fill0<<<2048, 256, 0, stream>>>(accP, (size_t)NN * 256);
            propagate256<<<(unsigned)(((size_t)NE * 32) / 256), 256, 0, stream>>>(
                cur, src, dst, nrm, accP);
            cast_f2b<<<(unsigned)(((size_t)NN * 256 / 4 + 255) / 256), 256, 0, stream>>>(
                accP, nxt, (size_t)NN * 256);
            bool last = (k == 3);
            gemm(nxt, 256, pT2 + k * 256 * 128, 256, last ? bT2 : nullptr, accT,
                 last ? nullptr : accT, 128, last ? h128 : nullptr, 128,
                 NN, 128, 256, 128, last ? 1 : 0);
            cur = nxt;
        }
    }
    concat2<<<(unsigned)(((size_t)NN * 256 + 255) / 256), 256, 0, stream>>>(
        xraw, h128, catb, NN, 128, 128);
    gemm(catb, 256, pW3, 256, b3, nullptr, nullptr, 0, a_bf, 128, NN, 128, 256, 128, 1);
    gemm(a_bf, 128, pW4, 128, b4, nullptr, outNemb, 128, nembB, 128, NN, 128, 128, 128, 0);

    // ---- stage 4: EdgeGLAM ----
    edge_bn_stats<<<272, 256, 0, stream>>>(outNemb, edge_raw, src, dst, meanE, varE);
    build_egin<<<(unsigned)(((size_t)NE * 288 + 255) / 256), 256, 0, stream>>>(
        outNemb, edge_raw, src, dst, meanE, varE, e_bn_g, e_bn_b, egmc);
    gemm(egmc, 288, pWe1, 288, be1, nullptr, nullptr, 0, eh, 128, NE, 128, 288, 128, 1);
    gemm(eh, 128, pWe2, 128, be2, nullptr, outEemb, 64, eembB, 64, NE, 64, 128, 64, 0);

    // ---- stage 5: category head ----
    build_mc<<<(unsigned)(((size_t)NE * 352 + 255) / 256), 256, 0, stream>>>(
        nembB, eembB, edge_raw, src, dst, egmc);
    gemm(egmc, 352, pWc1, 352, bc1, nullptr, nullptr, 0, ch1, 64, NE, 64, 352, 64, 1);
    gemm(ch1, 64, pWc2, 64, bc2, nullptr, nullptr, 0, ch2, 32, NE, 32, 64, 32, 1);
    gemm(ch2, 32, pWcl, 32, pbcl, nullptr, logit, 8, nullptr, 0, NE, 16, 32, 8, 0);
    softmax8<<<(NE + 255) / 256, 256, 0, stream>>>(logit, outMC);

    // ---- stage 6: binary head ----
    gemm(eembB, 64, pWb1, 64, bb1, nullptr, nullptr, 0, bh, 64, NE, 64, 64, 64, 1);
    gemm(bh, 64, pWb2, 64, pbb2, nullptr, outBin, 1, nullptr, 0, NE, 16, 64, 1, 2);

    (void)in_sizes; (void)out_size; (void)ws_size;
}
